// MixtureOfExperts_50964081934980
// MI455X (gfx1250) — compile-verified, matrix operand order checked by reading, not run on hardware
//
#include <hip/hip_runtime.h>
#include <cstdint>
#include <cstddef>

// ---------------------------------------------------------------------------
// MoE (B=8,S=4096,H=512,I=2048,E=8,K=2) for MI455X / gfx1250.
// Compute-bound (~275 GFLOP vs ~170MB HBM traffic) -> routed bf16 WMMA GEMMs.
// ROWS=64 token tiles, 512-thread blocks (16 waves). Wave tiling is
// tall-and-skinny: each wave owns few column tiles but all 64 rows, so every
// global B fragment feeds 4 wmmas (GEMM1) / 4 wmmas each of 2 frags (GEMM2),
// amortizing L2 latency structurally; B fragments are loaded by exactly one
// wave (zero duplicated global traffic), A fragments come from LDS.
// ---------------------------------------------------------------------------

#define BB 8
#define SS 4096
#define HH 512
#define II 2048
#define EE 8
#define NT (BB * SS)          /* 32768 tokens */
#define ROWS 64               /* token tile per block */
#define TILES_PER_E (NT / ROWS)

typedef __attribute__((ext_vector_type(16))) __bf16 bf16x16;
typedef __attribute__((ext_vector_type(8)))  float  floatx8;

union FragU { bf16x16 v; uint4 q[2]; };

__device__ __forceinline__ unsigned short f2bf(float f) {
    union { float f; unsigned u; } v; v.f = f;
    unsigned r = v.u + 0x7FFFu + ((v.u >> 16) & 1u);   // RNE
    return (unsigned short)(r >> 16);
}

// K-position held by (lane, element j) of a 16-bit WMMA operand fragment
// (per ISA 16-bit A-matrix 16x32 layout; B uses the same striping with N=lane&15).
__device__ __forceinline__ unsigned klocal(unsigned lane, unsigned j) {
    return (lane < 16u) ? ((j < 8u) ? j : j + 8u)
                        : ((j < 8u) ? j + 8u : j + 16u);
}

// ---------------- prep: x f32 -> bf16 ----------------
__global__ void cvt_x_kernel(const float* __restrict__ x,
                             unsigned short* __restrict__ xb) {
    size_t i = ((size_t)blockIdx.x * blockDim.x + threadIdx.x) * 4;
    float4 v = *(const float4*)(x + i);
    ushort4 o;
    o.x = f2bf(v.x); o.y = f2bf(v.y); o.z = f2bf(v.z); o.w = f2bf(v.w);
    *(ushort4*)(xb + i) = o;
}

// ---------------- prep: pack w1[E][H][I] into WMMA-B fragment-major bf16 ----
// dest index: (((e*128 + ctile)*16 + kstep)*32 + lane)*16 + j
__global__ void pack_w1_kernel(const float* __restrict__ w1,
                               unsigned short* __restrict__ w1p) {
    unsigned d = blockIdx.x * 256u + threadIdx.x;
    unsigned j = d & 15u, lane = (d >> 4) & 31u, s = (d >> 9) & 15u,
             c = (d >> 13) & 127u, e = d >> 20;
    unsigned k = s * 32u + klocal(lane, j);
    unsigned n = c * 16u + (lane & 15u);
    w1p[d] = f2bf(w1[((size_t)e * HH + k) * II + n]);
}

// ---------------- prep: pack w2[E][I][H] -----------------------------------
// dest index: (((e*32 + ctile)*64 + kstep)*32 + lane)*16 + j
__global__ void pack_w2_kernel(const float* __restrict__ w2,
                               unsigned short* __restrict__ w2p) {
    unsigned d = blockIdx.x * 256u + threadIdx.x;
    unsigned j = d & 15u, lane = (d >> 4) & 31u, s = (d >> 9) & 63u,
             c = (d >> 15) & 31u, e = d >> 20;
    unsigned k = s * 32u + klocal(lane, j);
    unsigned n = c * 16u + (lane & 15u);
    w2p[d] = f2bf(w2[((size_t)e * II + k) * HH + n]);
}

// ---------------- routing: one wave32 per token ----------------------------
__global__ void route_kernel(const float* __restrict__ x,
                             const float* __restrict__ gw,
                             const float* __restrict__ gb,
                             int* __restrict__ perm,
                             float* __restrict__ cwt,
                             int* __restrict__ counts) {
    int wid  = (int)((blockIdx.x * blockDim.x + threadIdx.x) >> 5);
    int lane = threadIdx.x & 31;
    if (wid >= NT) return;
    const float* xr = x + (size_t)wid * HH;

    float acc[EE];
#pragma unroll
    for (int e = 0; e < EE; ++e) acc[e] = 0.f;
    for (int h = lane; h < HH; h += 32) {
        float xv = xr[h];
        const float* g = gw + (size_t)h * EE;
#pragma unroll
        for (int e = 0; e < EE; ++e) acc[e] += xv * g[e];
    }
#pragma unroll
    for (int e = 0; e < EE; ++e) {
#pragma unroll
        for (int m = 16; m >= 1; m >>= 1) acc[e] += __shfl_xor(acc[e], m, 32);
    }
    if (lane == 0) {
        float l[EE], mx = -1e30f;
#pragma unroll
        for (int e = 0; e < EE; ++e) { l[e] = acc[e] + gb[e]; mx = l[e] > mx ? l[e] : mx; }
        float p[EE], sum = 0.f;
#pragma unroll
        for (int e = 0; e < EE; ++e) { p[e] = __expf(l[e] - mx); sum += p[e]; }
        int i0 = 0;
#pragma unroll
        for (int e = 1; e < EE; ++e) if (p[e] > p[i0]) i0 = e;
        int i1 = (i0 == 0) ? 1 : 0;
#pragma unroll
        for (int e = 0; e < EE; ++e) if (e != i0 && p[e] > p[i1]) i1 = e;
        float v0 = p[i0] / sum, v1 = p[i1] / sum;
        float e1 = __expf(v1 - v0);              // v1 <= v0, stable
        float w0 = 1.f / (1.f + e1), w1 = e1 / (1.f + e1);
        int pos0 = atomicAdd(&counts[i0], 1);
        perm[i0 * NT + pos0] = wid; cwt[i0 * NT + pos0] = w0;
        int pos1 = atomicAdd(&counts[i1], 1);
        perm[i1 * NT + pos1] = wid; cwt[i1 * NT + pos1] = w1;
    }
}

// ---------------- fused expert kernel --------------------------------------
// block = (tile of 64 routed tokens, expert). 512 threads = 16 waves.
// GEMM1: x[64x512] * w1[512x2048] -> relu -> h (bf16, LDS, 256-col chunks)
//        wave w owns I-tile (ic*16+w), all 64 rows: 1 B frag -> 4 wmma/k-step
// GEMM2: h[64x2048] * w2[2048x512] -> *ce -> atomicAdd out
//        wave w owns H-tiles {2w, 2w+1}, all 64 rows: 2 B frags -> 8 wmma/k-step
__global__ void __launch_bounds__(512, 1)
moe_kernel(const unsigned short* __restrict__ xb,
           const unsigned short* __restrict__ w1p,
           const unsigned short* __restrict__ w2p,
           const float* __restrict__ b1,
           const float* __restrict__ b2,
           const int* __restrict__ perm,
           const float* __restrict__ cwt,
           const int* __restrict__ counts,
           float* __restrict__ out) {
    const int e    = blockIdx.y;
    const int n    = counts[e];
    const int row0 = blockIdx.x * ROWS;
    if (row0 >= n) return;

    __shared__ __align__(16) unsigned short x_chunk[ROWS * 128]; // 16 KB
    __shared__ __align__(16) unsigned short h_chunk[ROWS * 256]; // 32 KB
    __shared__ int   tok_s[ROWS];
    __shared__ float ce_s[ROWS];

    const int tid  = threadIdx.x;
    const int wave = tid >> 5;         // 0..15
    const int lane = tid & 31;
    const int ln   = lane & 15;
    const int hi   = lane >> 4;

    if (tid < ROWS) {
        int r = row0 + tid;
        if (r < n) { tok_s[tid] = perm[e * NT + r]; ce_s[tid] = cwt[e * NT + r]; }
        else       { tok_s[tid] = 0;               ce_s[tid] = 0.f; }
    }
    __syncthreads();

    floatx8 oacc[8];                   // persistent GEMM2 accum: [stripe st][tile t] = st*2+t
#pragma unroll
    for (int t = 0; t < 8; ++t) oacc[t] = (floatx8)0.f;

    for (int ic = 0; ic < 8; ++ic) {   // I chunk of 256
        floatx8 hacc[4];               // GEMM1 accum: [stripe st]
#pragma unroll
        for (int t = 0; t < 4; ++t) hacc[t] = (floatx8)0.f;
        const int ct1 = ic * 16 + wave;            // this wave's I tile (of 128)

        for (int kc = 0; kc < 4; ++kc) {           // H chunk of 128
            __syncthreads();
            {   // stage gathered x rows [0..63], cols [kc*128, +128)
                int r = tid >> 3;                  // 0..63
                int part = (tid & 7) * 16;         // 16 elements = 32B
                const unsigned short* src =
                    xb + (size_t)tok_s[r] * HH + kc * 128 + part;
                const uint4* s4 = (const uint4*)src;
                uint4* dst = (uint4*)&x_chunk[r * 128 + part];
                dst[0] = s4[0]; dst[1] = s4[1];
                if (kc < 3) __builtin_prefetch(src + 128, 0, 1); // next K chunk
            }
            __syncthreads();
#pragma unroll
            for (int s = 0; s < 4; ++s) {          // k-step of 32
                int S1 = kc * 4 + s;               // global k-step 0..15
                FragU b;                           // single B frag, reused 4x
                const unsigned short* wbase =
                    w1p + ((((size_t)e * 128 + ct1) * 16 + S1) * 32 + lane) * 16;
                b.q[0] = *(const uint4*)(wbase);
                b.q[1] = *(const uint4*)(wbase + 8);
                int k0 = s * 32 + hi * 8;
                FragU a[4];
#pragma unroll
                for (int st = 0; st < 4; ++st) {   // A frags: 4 row stripes
                    int arow = st * 16 + ln;
                    a[st].q[0] = *(const uint4*)&x_chunk[arow * 128 + k0];
                    a[st].q[1] = *(const uint4*)&x_chunk[arow * 128 + k0 + 16];
                }
#pragma unroll
                for (int st = 0; st < 4; ++st) {
                    hacc[st] = __builtin_amdgcn_wmma_f32_16x16x32_bf16(
                        false, a[st].v, false, b.v, (short)0, hacc[st], false, false);
                }
            }
        }
        __syncthreads();
        // GEMM1 epilogue: bias + relu -> bf16 h_chunk (this wave's column tile)
        {
            int col  = ct1 * 16 + ln;              // global I column
            int lcol = wave * 16 + ln;             // 0..255
            float bv = b1[e * II + col];
#pragma unroll
            for (int st = 0; st < 4; ++st) {
#pragma unroll
                for (int r = 0; r < 8; ++r) {
                    int row = st * 16 + hi * 8 + r;
                    float v = hacc[st][r] + bv;
                    v = v > 0.f ? v : 0.f;
                    h_chunk[row * 256 + lcol] = f2bf(v);
                }
            }
        }
        __syncthreads();
        // GEMM2 partial over this I chunk (K=256 -> 8 k-steps)
#pragma unroll
        for (int s2 = 0; s2 < 8; ++s2) {
            int S2 = ic * 8 + s2;                  // global k-step 0..63
            FragU b[2];                            // 2 B frags, each reused 4x
            const unsigned short* wbase =
                w2p + ((((size_t)e * 32 + wave * 2) * 64 + S2) * 32 + lane) * 16;
#pragma unroll
            for (int t = 0; t < 2; ++t) {
                b[t].q[0] = *(const uint4*)(wbase + (size_t)t * (64 * 32 * 16));
                b[t].q[1] = *(const uint4*)(wbase + (size_t)t * (64 * 32 * 16) + 8);
            }
            int k0 = s2 * 32 + hi * 8;
            FragU a[4];
#pragma unroll
            for (int st = 0; st < 4; ++st) {       // A frags: 4 row stripes
                int arow = st * 16 + ln;
                a[st].q[0] = *(const uint4*)&h_chunk[arow * 256 + k0];
                a[st].q[1] = *(const uint4*)&h_chunk[arow * 256 + k0 + 16];
            }
#pragma unroll
            for (int st = 0; st < 4; ++st) {
#pragma unroll
                for (int t = 0; t < 2; ++t) {
                    oacc[st * 2 + t] = __builtin_amdgcn_wmma_f32_16x16x32_bf16(
                        false, a[st].v, false, b[t].v, (short)0, oacc[st * 2 + t],
                        false, false);
                }
            }
        }
    }

    // final epilogue: + b2, * combine weight, scatter-accumulate to out
#pragma unroll
    for (int t = 0; t < 2; ++t) {
        int col  = (wave * 2 + t) * 16 + ln;       // 0..511
        float b2v = b2[e * HH + col];
#pragma unroll
        for (int st = 0; st < 4; ++st) {
#pragma unroll
            for (int r = 0; r < 8; ++r) {
                int row = st * 16 + hi * 8 + r;
                float v = (oacc[st * 2 + t][r] + b2v) * ce_s[row];
                atomicAdd(out + (size_t)tok_s[row] * HH + col, v);
            }
        }
    }
}

// ---------------------------------------------------------------------------
extern "C" void kernel_launch(void* const* d_in, const int* in_sizes, int n_in,
                              void* d_out, int out_size, void* d_ws, size_t ws_size,
                              hipStream_t stream) {
    (void)in_sizes; (void)n_in; (void)out_size; (void)ws_size;
    const float* x  = (const float*)d_in[0];
    const float* gw = (const float*)d_in[1];
    const float* gb = (const float*)d_in[2];
    const float* w1 = (const float*)d_in[3];
    const float* b1 = (const float*)d_in[4];
    const float* w2 = (const float*)d_in[5];
    const float* b2 = (const float*)d_in[6];
    float* out = (float*)d_out;

    char* ws = (char*)d_ws;
    size_t off = 0;
    int* counts = (int*)(ws + off);               off += 256;
    unsigned short* xb  = (unsigned short*)(ws + off); off += (size_t)NT * HH * 2;       // 33.5 MB
    unsigned short* w1p = (unsigned short*)(ws + off); off += (size_t)EE * HH * II * 2;  // 16.8 MB
    unsigned short* w2p = (unsigned short*)(ws + off); off += (size_t)EE * II * HH * 2;  // 16.8 MB
    int*   perm = (int*)(ws + off);               off += (size_t)EE * NT * 4;            // 1 MB
    float* cwt  = (float*)(ws + off);             off += (size_t)EE * NT * 4;            // 1 MB

    hipMemsetAsync(counts, 0, 256, stream);
    hipMemsetAsync(out, 0, (size_t)NT * HH * sizeof(float), stream);

    cvt_x_kernel  <<<(NT * HH / 4) / 256, 256, 0, stream>>>(x, xb);
    pack_w1_kernel<<<(EE * HH * II) / 256, 256, 0, stream>>>(w1, w1p);
    pack_w2_kernel<<<(EE * II * HH) / 256, 256, 0, stream>>>(w2, w2p);
    route_kernel  <<<NT / 8, 256, 0, stream>>>(x, gw, gb, perm, cwt, counts);

    dim3 grid(TILES_PER_E, EE);
    moe_kernel<<<grid, 512, 0, stream>>>(xb, w1p, w2p, b1, b2, perm, cwt, counts, out);
}
